// VanillaRNN_4518305595514
// MI455X (gfx1250) — matrix operand order repaired and checked by reading, not compiled, not run on hardware
//
#include <hip/hip_runtime.h>
#include <hip/hip_bf16.h>

typedef __attribute__((ext_vector_type(16))) __bf16 v16bf;
typedef __attribute__((ext_vector_type(8)))  float  v8f;

#define DIM 512
#define SEQ 1024
#define NC  10

static __device__ __forceinline__ unsigned short f2bf(float f) {
    unsigned int u = __float_as_uint(f);
    u += 0x7FFFu + ((u >> 16) & 1u);            // round-to-nearest-even
    return (unsigned short)(u >> 16);
}
static __device__ __forceinline__ float bf2f(unsigned short h) {
    return __uint_as_float(((unsigned int)h) << 16);
}

// ---- prep: f32 -> bf16 conversion of Whh (512x512) ----
__global__ void k_f32_to_bf16(const float* __restrict__ src,
                              unsigned short* __restrict__ dst, int n4) {
    int id = blockIdx.x * blockDim.x + threadIdx.x;
    if (id >= n4) return;
    float4 v = ((const float4*)src)[id];
    unsigned int lo = (unsigned int)f2bf(v.x) | ((unsigned int)f2bf(v.y) << 16);
    unsigned int hi = (unsigned int)f2bf(v.z) | ((unsigned int)f2bf(v.w) << 16);
    ((uint2*)dst)[id] = make_uint2(lo, hi);
}

// ---- prep: zero the initial hT state (bf16) ----
__global__ void k_zero16(uint4* __restrict__ dst, int n) {
    int id = blockIdx.x * blockDim.x + threadIdx.x;
    if (id < n) dst[id] = make_uint4(0u, 0u, 0u, 0u);
}

// ---- prep: U[t][i] = (Whx @ x[:,t])[i] for all timesteps ----
__global__ void k_precompute_u(const float* __restrict__ Whx,
                               const float* __restrict__ x,
                               float* __restrict__ U) {
    const int t = blockIdx.x;
    for (int i = threadIdx.x; i < DIM; i += blockDim.x) {
        const float* wrow = Whx + i * DIM;
        float acc = 0.f;
        for (int k = 0; k < DIM; ++k)
            acc = fmaf(wrow[k], x[k * SEQ + t], acc);
        U[t * DIM + i] = acc;
    }
}

// ---- one RNN step: hT_out[j][i] = tanh(U[t][i] + sum_k Whh[i][k]*hT_in[j][k] + bh[j])
// A = Whh (bf16, row-major i x k), B = hT_in (bf16, row-major j x k).
// Wave tile: 32(M) x 16(N), two 16x16 accumulators sharing one B fragment.
// Workgroup: 8 waves as 2(M) x 4(N) -> 64x64 tile. Grid (8,8).
__global__ __launch_bounds__(256) void k_rnn_step(
    const unsigned short* __restrict__ A,
    const unsigned short* __restrict__ B,
    unsigned short* __restrict__ Hout,
    const float* __restrict__ U,
    const float* __restrict__ bh,
    int t)
{
    const int tid  = threadIdx.x;
    const int wave = tid >> 5;
    const int lane = tid & 31;
    const int hi   = lane >> 4;      // half-wave select
    const int lr   = lane & 15;
    const int wm   = wave >> 2;      // 0..1
    const int wn   = wave & 3;       // 0..3
    const int mwave = blockIdx.x * 64 + wm * 32;            // wave M base
    const int j     = blockIdx.y * 64 + wn * 16 + lr;       // this lane's column

    union Frag { uint4 u[2]; v16bf v; };

    v8f acc[2] = {};

    const uint4* brow = (const uint4*)(B + j * DIM);        // row of hT (K contiguous)
    // kc counts uint4 units (8 bf16 each); 32 K-elems (4 uint4) per iteration
    for (int kc = 0; kc < DIM / 8; kc += 4) {
        Frag fb;
        // B 32x16 layout: lanes 0-15: K=0..7 & 16..23; lanes 16-31: K=8..15 & 24..31
        fb.u[0] = brow[kc + hi];
        fb.u[1] = brow[kc + hi + 2];
        #pragma unroll
        for (int s = 0; s < 2; ++s) {
            const uint4* arow = (const uint4*)(A + (mwave + s * 16 + lr) * DIM);
            Frag fa;
            // A 16x32 layout: lanes 0-15: K=0..7 & 16..23; lanes 16-31: K=8..15 & 24..31
            fa.u[0] = arow[kc + hi];
            fa.u[1] = arow[kc + hi + 2];
            acc[s] = __builtin_amdgcn_wmma_f32_16x16x32_bf16(
                false, fa.v, false, fb.v, (short)0, acc[s], false, false);
        }
    }

    // Epilogue: C layout -> lane's 8 accum elements are 8 consecutive rows i
    // at column j; store tanh(...) as 8 packed bf16 (one b128) in transposed layout.
    const float  bhv = bh[j];
    const float* ut  = U + t * DIM;
    #pragma unroll
    for (int s = 0; s < 2; ++s) {
        const int i0 = mwave + s * 16 + 8 * hi;
        unsigned int w[4];
        #pragma unroll
        for (int p = 0; p < 4; ++p) {
            float v0 = tanhf(acc[s][2 * p]     + ut[i0 + 2 * p]     + bhv);
            float v1 = tanhf(acc[s][2 * p + 1] + ut[i0 + 2 * p + 1] + bhv);
            w[p] = (unsigned int)f2bf(v0) | ((unsigned int)f2bf(v1) << 16);
        }
        *(uint4*)(Hout + j * DIM + i0) = make_uint4(w[0], w[1], w[2], w[3]);
    }
}

// ---- head: p[i][c] = softmax_c( sum_j hT[j][i]*Wph[j][c] + bp[c] ) ----
__global__ void k_head(const unsigned short* __restrict__ H,
                       const float* __restrict__ Wph,
                       const float* __restrict__ bp,
                       float* __restrict__ out) {
    int i = blockIdx.x * blockDim.x + threadIdx.x;
    if (i >= DIM) return;
    float acc[NC];
    #pragma unroll
    for (int c = 0; c < NC; ++c) acc[c] = bp[c];
    for (int jj = 0; jj < DIM; ++jj) {
        float hv = bf2f(H[jj * DIM + i]);    // coalesced across i
        const float* wr = Wph + jj * NC;
        #pragma unroll
        for (int c = 0; c < NC; ++c) acc[c] = fmaf(hv, wr[c], acc[c]);
    }
    float m = acc[0];
    #pragma unroll
    for (int c = 1; c < NC; ++c) m = fmaxf(m, acc[c]);
    float sum = 0.f;
    #pragma unroll
    for (int c = 0; c < NC; ++c) { acc[c] = expf(acc[c] - m); sum += acc[c]; }
    float inv = 1.f / sum;
    #pragma unroll
    for (int c = 0; c < NC; ++c) out[i * NC + c] = acc[c] * inv;
}

extern "C" void kernel_launch(void* const* d_in, const int* in_sizes, int n_in,
                              void* d_out, int out_size, void* d_ws, size_t ws_size,
                              hipStream_t stream) {
    const float* x   = (const float*)d_in[0];
    const float* Whx = (const float*)d_in[1];
    const float* Whh = (const float*)d_in[2];
    const float* Wph = (const float*)d_in[3];
    const float* bh  = (const float*)d_in[4];
    const float* bp  = (const float*)d_in[5];
    float* out = (float*)d_out;

    const size_t HBYTES = (size_t)DIM * DIM * sizeof(unsigned short); // 512 KB
    char* ws = (char*)d_ws;
    unsigned short* whh_bf = (unsigned short*)(ws);
    unsigned short* h0     = (unsigned short*)(ws + HBYTES);
    unsigned short* h1     = (unsigned short*)(ws + 2 * HBYTES);
    float*          U      = (float*)(ws + 3 * HBYTES);               // 2 MB

    {   // Whh -> bf16
        int n4 = DIM * DIM / 4;
        k_f32_to_bf16<<<(n4 + 255) / 256, 256, 0, stream>>>(Whh, whh_bf, n4);
    }
    {   // hT(0) = 0
        int n16 = (int)(HBYTES / 16);
        k_zero16<<<(n16 + 255) / 256, 256, 0, stream>>>((uint4*)h0, n16);
    }
    // U[t][i] for all timesteps
    k_precompute_u<<<SEQ, 256, 0, stream>>>(Whx, x, U);

    unsigned short* hin  = h0;
    unsigned short* hout = h1;
    for (int t = 0; t < SEQ; ++t) {
        k_rnn_step<<<dim3(8, 8), 256, 0, stream>>>(whh_bf, hin, hout, U, bh, t);
        unsigned short* tmp = hin; hin = hout; hout = tmp;
    }
    // after 1024 steps, last state is back in h0 (== hin)
    k_head<<<(DIM + 255) / 256, 256, 0, stream>>>(hin, Wph, bp, out);
}
